// OrthogonalCell_57913339019729
// MI455X (gfx1250) — matrix operand order repaired and checked by reading, not compiled, not run on hardware
//
#include <hip/hip_runtime.h>
#include <hip/hip_bf16.h>
#include <math.h>

// Problem constants (match reference)
#define B_      64
#define T_      1024
#define D_      128
#define UNITS_  512
#define ORDER_  256

// Persistent-kernel shape: 8 WGs x 16 wave32.
// WG id = (role<<2) | mblock : mblock owns activation rows [16*mb, 16*mb+16).
// role 0: X-phase + h-columns 0..255 ; role 1: M-phase + h-columns 256..511.
#define NWG 8
#define TPB 512

// LDS row padding: +8 bf16 (16B) => row stride 1040B (s_h) == 4 dwords mod 64
// banks -> the 16-row x b128 A-tile read hits all 64 banks conflict-free.
#define PAD 8
#define LDH (UNITS_ + PAD)
#define LDX (ORDER_ + PAD)
#define LDD (D_ + PAD)

typedef __attribute__((ext_vector_type(16))) __bf16       v16bf;
typedef __attribute__((ext_vector_type(8)))  float        v8f;
typedef __attribute__((ext_vector_type(4)))  unsigned int v4u;

union ABits { v4u q[2]; v16bf v; };
static_assert(sizeof(v16bf) == 32, "v16bf must be 32B");

// ---------------------------------------------------------------------------
// A-operand tile load (16x32 bf16) from row-major GLOBAL buffer [rows][ld].
// CDNA5 A layout: lanes 0-15 hold K {k0..k0+7, k0+16..k0+23}, lanes 16-31
// hold K {k0+8..k0+15, k0+24..k0+31}; 2 bf16/VGPR => two b128 loads.
// ---------------------------------------------------------------------------
__device__ __forceinline__ v16bf load_a_glb(const __hip_bfloat16* __restrict__ P,
                                            int ld, int m0, int k0, int lane) {
  int row = m0 + (lane & 15);
  int kb  = k0 + ((lane >> 4) << 3);
  const __hip_bfloat16* p = P + row * ld + kb;
  ABits u;
  u.q[0] = *(const v4u*)(p);
  u.q[1] = *(const v4u*)(p + 16);
  return u.v;
}

// Same A-tile fetch but from a 16-row LDS staging buffer (padded stride ldp).
__device__ __forceinline__ v16bf load_a_lds(const __hip_bfloat16* S,
                                            int ldp, int k0, int lane) {
  int row = lane & 15;
  int kb  = k0 + ((lane >> 4) << 3);
  const __hip_bfloat16* p = S + row * ldp + kb;
  ABits u;
  u.q[0] = *(const v4u*)(p);
  u.q[1] = *(const v4u*)(p + 16);
  return u.v;
}

// B-operand tile: weights pre-swizzled so each 32x16 tile is 256 dwords with
// each lane's 8 dwords contiguous => wave reads 1KB contiguous (L2-resident).
__device__ __forceinline__ v16bf load_b_tile(const unsigned int* __restrict__ W,
                                             int tileIdx, int lane) {
  const unsigned int* p = W + tileIdx * 256 + lane * 8;
  ABits u;
  u.q[0] = *(const v4u*)(p);
  u.q[1] = *(const v4u*)(p + 4);
  return u.v;
}

__device__ __forceinline__ v8f wmma_bf16(v16bf a, v16bf b, v8f c) {
  return __builtin_amdgcn_wmma_f32_16x16x32_bf16(false, a, false, b,
                                                 (short)0, c, false, false);
}

// Cooperative copy of 16 rows x ld bf16 (row-major global) into padded LDS.
// nvr = ld/8 is a power of two at every call site -> divisions become shifts.
__device__ __forceinline__ void stage_rows(__hip_bfloat16* __restrict__ dst, int ldp,
                                           const __hip_bfloat16* __restrict__ src, int ld) {
  const int nvr = ld >> 3;                 // 16B vectors per row
  for (int i = threadIdx.x; i < 16 * nvr; i += TPB) {
    int r = i / nvr, v = i - r * nvr;
    *(v4u*)(dst + r * ldp + (v << 3)) = *(const v4u*)(src + r * ld + (v << 3));
  }
}

// Incrementing-ticket device-wide barrier.
__device__ __forceinline__ void grid_barrier(unsigned int* bar, unsigned int target) {
  __syncthreads();
  if (threadIdx.x == 0) {
    __threadfence();
    atomicAdd(bar, 1u);
    while (__hip_atomic_load(bar, __ATOMIC_RELAXED, __HIP_MEMORY_SCOPE_AGENT) < target)
      __builtin_amdgcn_s_sleep(1);
    __threadfence();
  }
  __syncthreads();
}

// ---------------------------------------------------------------------------
// Prep kernel 1: swizzle fp32 row-major [K][N] weight into bf16 WMMA-B tile
// order (tileIdx = nt*(K/32)+kt; lane = 16*((k%16)/8)+n%16; 2 bf16/VGPR).
// ---------------------------------------------------------------------------
__global__ void swizzle_w_kernel(const float* __restrict__ src,
                                 __hip_bfloat16* __restrict__ dst, int K, int N) {
  int idx = blockIdx.x * blockDim.x + threadIdx.x;
  if (idx >= K * N) return;
  int k = idx / N, n = idx % N;
  int kt = k >> 5, nt = n >> 4;
  int kk = k & 31, nn = n & 15;
  int lane = (((kk >> 3) & 1) << 4) | nn;
  int j    = ((kk >> 4) << 2) | ((kk & 7) >> 1);
  int half = kk & 1;
  int tileIdx = nt * (K >> 5) + kt;
  int pos = tileIdx * 512 + lane * 16 + j * 2 + half;
  dst[pos] = __float2bfloat16(src[k * N + n]);
}

// Prep kernel 2: zero initial states, u0 = inputs[:,0,:], reset barrier ticket.
__global__ void init_state_kernel(const float* __restrict__ inp,
                                  __hip_bfloat16* __restrict__ hS0,
                                  __hip_bfloat16* __restrict__ xS0,
                                  __hip_bfloat16* __restrict__ uS,
                                  unsigned int* __restrict__ bar) {
  int idx = blockIdx.x * blockDim.x + threadIdx.x;
  if (idx == 0) *bar = 0u;
  if (idx < B_ * UNITS_) hS0[idx] = __float2bfloat16(0.0f);
  if (idx < B_ * ORDER_) xS0[idx] = __float2bfloat16(0.0f);
  if (idx < B_ * D_) {
    int b = idx / D_, d = idx % D_;
    uS[idx] = __float2bfloat16(inp[(size_t)b * T_ * D_ + d]);
  }
}

// ---------------------------------------------------------------------------
// Persistent RNN kernel. Per step (2 device barriers):
//   region1: stage h_{t-1}, x_t to LDS | role1: M_t | all: partial H (24 wmma)
//   -- barrier (publishes f_t, u_{t+1}) --
//   region2: stage f_t, u_{t+1} | all: finish H (4 wmma) + tanh + store
//            role0: X_{t+1} (12 wmma)
//   -- barrier --
// ---------------------------------------------------------------------------
__global__ __launch_bounds__(TPB, 1)
void lrnn_persistent_kernel(const float* __restrict__ inp,
                            const float* __restrict__ wfm,
                            const float* __restrict__ by,
                            const unsigned int* __restrict__ wyh,
                            const unsigned int* __restrict__ wyx,
                            const unsigned int* __restrict__ wyf,
                            const unsigned int* __restrict__ atw,
                            const unsigned int* __restrict__ btw,
                            const unsigned int* __restrict__ ctw,
                            __hip_bfloat16* __restrict__ hS,   // 2 x [64][512]
                            __hip_bfloat16* __restrict__ xS,   // 2 x [64][256]
                            __hip_bfloat16* __restrict__ fS,   // [64][128]
                            __hip_bfloat16* __restrict__ uS,   // [64][128]
                            float* __restrict__ out,           // [64][1024][512]
                            unsigned int* __restrict__ bar) {
  __shared__ __align__(16) unsigned short s_h_raw[16 * LDH];
  __shared__ __align__(16) unsigned short s_x_raw[16 * LDX];
  __shared__ __align__(16) unsigned short s_f_raw[16 * LDD];
  __shared__ __align__(16) unsigned short s_u_raw[16 * LDD];
  __hip_bfloat16* s_h = (__hip_bfloat16*)s_h_raw;
  __hip_bfloat16* s_x = (__hip_bfloat16*)s_x_raw;
  __hip_bfloat16* s_f = (__hip_bfloat16*)s_f_raw;
  __hip_bfloat16* s_u = (__hip_bfloat16*)s_u_raw;

  const int lane = threadIdx.x & 31;
  const int wave = threadIdx.x >> 5;
  const int mb   = blockIdx.x & 3;          // row block: rows [m0, m0+16)
  const int role = blockIdx.x >> 2;         // 0: X + h-cols lo, 1: M + h-cols hi
  const int m0   = mb << 4;

  const int  h_nt = (role << 4) + wave;     // 0..31
  const bool do_x = (role == 0);
  const int  x_nt = wave;                   // 0..15
  const bool do_m = (role == 1) && (wave < 8);
  const int  m_nt = wave;                   // 0..7

  const int   colh = (h_nt << 4) + (lane & 15);
  const int   colx = (x_nt << 4) + (lane & 15);
  const int   colm = (m_nt << 4) + (lane & 15);
  const int   rb   = m0 + ((lane >> 4) << 3);
  const float bias = by[colh];
  const float wf   = do_m ? wfm[colm] : 0.0f;

  unsigned int tgt = 0;

  // ---- prologue: x_0 = 0@AT + u_0@BT (global path; xS buf0 is zeros)
  if (do_x) {
    v8f acc = {0.f, 0.f, 0.f, 0.f, 0.f, 0.f, 0.f, 0.f};
#pragma unroll
    for (int kt = 0; kt < ORDER_ / 32; ++kt)
      acc = wmma_bf16(load_a_glb(xS, ORDER_, m0, kt * 32, lane),
                      load_b_tile(atw, x_nt * (ORDER_ / 32) + kt, lane), acc);
#pragma unroll
    for (int kt = 0; kt < D_ / 32; ++kt)
      acc = wmma_bf16(load_a_glb(uS, D_, m0, kt * 32, lane),
                      load_b_tile(btw, x_nt * (D_ / 32) + kt, lane), acc);
    __hip_bfloat16* x1 = xS + B_ * ORDER_;
#pragma unroll
    for (int r = 0; r < 8; ++r)
      x1[(rb + r) * ORDER_ + colx] = __float2bfloat16(acc[r]);
  }
  tgt += NWG; grid_barrier(bar, tgt);

  for (int t = 0; t < T_; ++t) {
    const int cur = t & 1, nxt = cur ^ 1;
    const __hip_bfloat16* hcur = hS + cur * (B_ * UNITS_);
    __hip_bfloat16*       hnxt = hS + nxt * (B_ * UNITS_);
    const __hip_bfloat16* xt   = xS + nxt * (B_ * ORDER_);  // x_t
    __hip_bfloat16*       xnew = xS + cur * (B_ * ORDER_);  // slot for x_{t+1}

    // ======== region 1: stage h,x ; M_t ; partial H ========
    stage_rows(s_h, LDH, hcur + m0 * UNITS_, UNITS_);
    stage_rows(s_x, LDX, xt + m0 * ORDER_, ORDER_);
    __syncthreads();

    if (do_m) {
      v8f accM = {0.f, 0.f, 0.f, 0.f, 0.f, 0.f, 0.f, 0.f};
#pragma unroll
      for (int kt = 0; kt < ORDER_ / 32; ++kt)
        accM = wmma_bf16(load_a_lds(s_x, LDX, kt * 32, lane),
                         load_b_tile(ctw, m_nt * (ORDER_ / 32) + kt, lane), accM);
#pragma unroll
      for (int r = 0; r < 8; ++r) {
        const float mv = accM[r];
        fS[(rb + r) * D_ + colm] = __float2bfloat16(wf * mv);
        float un = 0.0f;
        if (t + 1 < T_)
          un = inp[((size_t)(rb + r) * T_ + (t + 1)) * D_ + colm] + mv;
        uS[(rb + r) * D_ + colm] = __float2bfloat16(un);
      }
    }

    // partial H: h_{t-1}@WyhT + x_t@WyxT + by (does not need f_t)
    v8f accH;
#pragma unroll
    for (int r = 0; r < 8; ++r) accH[r] = bias;
#pragma unroll
    for (int kt = 0; kt < UNITS_ / 32; ++kt)
      accH = wmma_bf16(load_a_lds(s_h, LDH, kt * 32, lane),
                       load_b_tile(wyh, h_nt * (UNITS_ / 32) + kt, lane), accH);
#pragma unroll
    for (int kt = 0; kt < ORDER_ / 32; ++kt)
      accH = wmma_bf16(load_a_lds(s_x, LDX, kt * 32, lane),
                       load_b_tile(wyx, h_nt * (ORDER_ / 32) + kt, lane), accH);

    tgt += NWG; grid_barrier(bar, tgt);   // publishes f_t, u_{t+1}

    // ======== region 2: stage f,u ; finish H ; X_{t+1} ========
    stage_rows(s_f, LDD, fS + m0 * D_, D_);
    if (do_x) stage_rows(s_u, LDD, uS + m0 * D_, D_);
    __syncthreads();

#pragma unroll
    for (int kt = 0; kt < D_ / 32; ++kt)
      accH = wmma_bf16(load_a_lds(s_f, LDD, kt * 32, lane),
                       load_b_tile(wyf, h_nt * (D_ / 32) + kt, lane), accH);
#pragma unroll
    for (int r = 0; r < 8; ++r) {
      const float hv = tanhf(accH[r]);
      out[((size_t)(rb + r) * T_ + t) * UNITS_ + colh] = hv;   // fp32 output
      hnxt[(rb + r) * UNITS_ + colh] = __float2bfloat16(hv);   // bf16 state
    }

    if (do_x && (t + 1 < T_)) {           // x_{t+1} = x_t@AT + u_{t+1}@BT
      v8f accX = {0.f, 0.f, 0.f, 0.f, 0.f, 0.f, 0.f, 0.f};
#pragma unroll
      for (int kt = 0; kt < ORDER_ / 32; ++kt)
        accX = wmma_bf16(load_a_lds(s_x, LDX, kt * 32, lane),
                         load_b_tile(atw, x_nt * (ORDER_ / 32) + kt, lane), accX);
#pragma unroll
      for (int kt = 0; kt < D_ / 32; ++kt)
        accX = wmma_bf16(load_a_lds(s_u, LDD, kt * 32, lane),
                         load_b_tile(btw, x_nt * (D_ / 32) + kt, lane), accX);
#pragma unroll
      for (int r = 0; r < 8; ++r)
        xnew[(rb + r) * ORDER_ + colx] = __float2bfloat16(accX[r]);
    }

    tgt += NWG; grid_barrier(bar, tgt);
  }
}

// ---------------------------------------------------------------------------
// Host launcher
// ---------------------------------------------------------------------------
extern "C" void kernel_launch(void* const* d_in, const int* in_sizes, int n_in,
                              void* d_out, int out_size, void* d_ws, size_t ws_size,
                              hipStream_t stream) {
  (void)in_sizes; (void)n_in; (void)out_size; (void)ws_size;

  const float* inp   = (const float*)d_in[0];  // [64][1024][128]
  const float* wfm   = (const float*)d_in[1];  // [128]
  const float* wyh_f = (const float*)d_in[2];  // [512][512]
  const float* wyx_f = (const float*)d_in[3];  // [256][512]
  const float* wyf_f = (const float*)d_in[4];  // [128][512]
  const float* by    = (const float*)d_in[5];  // [512]
  const float* at_f  = (const float*)d_in[6];  // [256][256]
  const float* bt_f  = (const float*)d_in[7];  // [128][256]
  const float* ct_f  = (const float*)d_in[8];  // [256][128]
  float* out = (float*)d_out;

  char* ws = (char*)d_ws;
  size_t off = 0;
  auto take = [&](size_t bytes) -> char* {
    char* p = ws + off;
    off += (bytes + 255) & ~(size_t)255;
    return p;
  };
  __hip_bfloat16* wyh = (__hip_bfloat16*)take((size_t)512 * 512 * 2);
  __hip_bfloat16* wyx = (__hip_bfloat16*)take((size_t)256 * 512 * 2);
  __hip_bfloat16* wyf = (__hip_bfloat16*)take((size_t)128 * 512 * 2);
  __hip_bfloat16* atw = (__hip_bfloat16*)take((size_t)256 * 256 * 2);
  __hip_bfloat16* btw = (__hip_bfloat16*)take((size_t)128 * 256 * 2);
  __hip_bfloat16* ctw = (__hip_bfloat16*)take((size_t)256 * 128 * 2);
  __hip_bfloat16* hS  = (__hip_bfloat16*)take((size_t)2 * B_ * UNITS_ * 2);
  __hip_bfloat16* xS  = (__hip_bfloat16*)take((size_t)2 * B_ * ORDER_ * 2);
  __hip_bfloat16* fS  = (__hip_bfloat16*)take((size_t)B_ * D_ * 2);
  __hip_bfloat16* uS  = (__hip_bfloat16*)take((size_t)B_ * D_ * 2);
  unsigned int*   bar = (unsigned int*)take(256);

  const int PT = 256;
  auto blocks = [](int n, int t) { return (n + t - 1) / t; };
  swizzle_w_kernel<<<blocks(512 * 512, PT), PT, 0, stream>>>(wyh_f, wyh, 512, 512);
  swizzle_w_kernel<<<blocks(256 * 512, PT), PT, 0, stream>>>(wyx_f, wyx, 256, 512);
  swizzle_w_kernel<<<blocks(128 * 512, PT), PT, 0, stream>>>(wyf_f, wyf, 128, 512);
  swizzle_w_kernel<<<blocks(256 * 256, PT), PT, 0, stream>>>(at_f, atw, 256, 256);
  swizzle_w_kernel<<<blocks(128 * 256, PT), PT, 0, stream>>>(bt_f, btw, 128, 256);
  swizzle_w_kernel<<<blocks(256 * 128, PT), PT, 0, stream>>>(ct_f, ctw, 256, 128);
  init_state_kernel<<<blocks(B_ * UNITS_, PT), PT, 0, stream>>>(inp, hS, xS, uS, bar);

  lrnn_persistent_kernel<<<NWG, TPB, 0, stream>>>(
      inp, wfm, by,
      (const unsigned int*)wyh, (const unsigned int*)wyx, (const unsigned int*)wyf,
      (const unsigned int*)atw, (const unsigned int*)btw, (const unsigned int*)ctw,
      hS, xS, fS, uS, out, bar);
}
